// HGC_ALL_25237227832006
// MI455X (gfx1250) — matrix operand order repaired and checked by reading, not compiled, not run on hardware
//
#include <hip/hip_runtime.h>
#include <hip/hip_bf16.h>

// ---------------------------------------------------------------------------
// CDNA5 (gfx1250) WMMA types
// ---------------------------------------------------------------------------
typedef __attribute__((ext_vector_type(16))) _Float16 v16h;
typedef __attribute__((ext_vector_type(4)))  _Float16 v4h;
typedef __attribute__((ext_vector_type(8)))  float    v8f;

#define DD    128
#define CCOL  2000
#define CPAD  2048
#define NLRN  20000
#define NSCN  2000
#define NCPT  8000
// NLRN, NCPT are multiples of 32 (TM=2); NSCN is a multiple of 16 (TM=1).

// ---------------------------------------------------------------------------
// tableT[d][j] = (relu(j*W1 + b1) @ W2 + b2)[d], f16, K padded to CPAD=2048.
// ---------------------------------------------------------------------------
__global__ __launch_bounds__(128)
void build_table_kernel(const float* __restrict__ W1, const float* __restrict__ b1,
                        const float* __restrict__ W2, const float* __restrict__ b2,
                        _Float16* __restrict__ tableT)
{
    int j = blockIdx.x;
    int d = threadIdx.x;
    float v = 0.0f;
    if (j < CCOL) {
        float jf = (float)j;
#pragma unroll
        for (int k = 0; k < 16; ++k) {
            float hk = jf * W1[k] + b1[k];
            hk = hk > 0.0f ? hk : 0.0f;
            v += hk * W2[k * DD + d];
        }
        v += b2[d];
    }
    tableT[(size_t)d * CPAD + j] = (_Float16)v;
}

// Wt[n][k] = W[k][n] in f16 (contiguous in K per output column -> B fragment).
__global__ __launch_bounds__(128)
void wt_kernel(const float* __restrict__ W, _Float16* __restrict__ Wt)
{
    int n = blockIdx.x;
    int k = threadIdx.x;
    Wt[(size_t)n * DD + k] = (_Float16)W[(size_t)k * DD + n];
}

// ---------------------------------------------------------------------------
// Cooperative A staging: 16*TM rows x 32 K of mask -> {0,1} f16 fragments.
// Thread t handles one 4-K run: one float4 global load + one 8B LDS store.
// Fragment layout: lane = m + 16*(K>>4), half = K&15;
// a 4-run (K4 aligned) is contiguous in halves.
// ---------------------------------------------------------------------------
template <int TM, bool KG>
__device__ __forceinline__
void proj_stage(const float* __restrict__ mat, int row0, int k0, int tid,
                _Float16* __restrict__ aT)
{
    if (tid < 128 * TM) {
        int mrow = tid >> 3;              // 0 .. 16*TM-1
        int kk4  = (tid & 7) * 4;         // 0,4,...,28
        int gk   = k0 + kk4;
        const float* rp = mat + (size_t)(row0 + mrow) * CCOL + gk;
        float4 x;
        if (KG) {
            x = (gk + 4 <= CCOL) ? *(const float4*)rp : make_float4(0.f, 0.f, 0.f, 0.f);
        } else {
            x = *(const float4*)rp;
            __builtin_prefetch(rp + 256, 0, 0);     // global_prefetch_b8, ~8 tiles ahead
        }
        int tt = mrow >> 4;
        int m  = mrow & 15;
        int lt = m + 16 * (kk4 >> 4);
        int hb = kk4 & 15;
        v4h c;
        c[0] = (_Float16)((x.x != 0.f) ? 1.f : 0.f);
        c[1] = (_Float16)((x.y != 0.f) ? 1.f : 0.f);
        c[2] = (_Float16)((x.z != 0.f) ? 1.f : 0.f);
        c[3] = (_Float16)((x.w != 0.f) ? 1.f : 0.f);
        *(v4h*)(aT + tt * 512 + lt * 16 + hb) = c;
    }
}

// ---------------------------------------------------------------------------
// Projection GEMM + fused masked-mean: emb[N,128] = maskedmean((mat!=0)@table).
// 256 thr = 8 waves; wave w owns cols [16w,16w+16); 16*TM rows per block.
// One B fragment feeds 2*TM WMMAs (data + exact row-count vs all-ones B).
// Double-buffered LDS A tile -> single barrier per K iteration.
// ---------------------------------------------------------------------------
template <int TM>
__global__ __launch_bounds__(256)
void proj_gemm_kernel(const float* __restrict__ mat, const _Float16* __restrict__ tableT,
                      float* __restrict__ emb)
{
    __shared__ __align__(32) _Float16 aTile[2][TM * 512];
    int tid  = threadIdx.x;
    int wave = tid >> 5;
    int lane = tid & 31;
    int row0 = blockIdx.x * (16 * TM);
    int col0 = wave * 16;
    int m    = lane & 15;
    int hi   = lane >> 4;
    const _Float16* bcol = tableT + (size_t)(col0 + m) * CPAD + hi * 16;

    v16h onesb;
#pragma unroll
    for (int h = 0; h < 16; ++h) onesb[h] = (_Float16)1.0f;

    v8f acc[TM]  = {};
    v8f cacc[TM] = {};

    int k0 = 0, p = 0;
    for (; k0 + 32 <= CCOL; k0 += 32, p ^= 1) {
        proj_stage<TM, false>(mat, row0, k0, tid, aTile[p]);
        __syncthreads();
        v16h b = *(const v16h*)(bcol + k0);
#pragma unroll
        for (int tt = 0; tt < TM; ++tt) {
            v16h a = *(const v16h*)(aTile[p] + tt * 512 + lane * 16);
            acc[tt]  = __builtin_amdgcn_wmma_f32_16x16x32_f16(false, a, false, b,
                                                              (short)0, acc[tt],  false, false);
            cacc[tt] = __builtin_amdgcn_wmma_f32_16x16x32_f16(false, a, false, onesb,
                                                              (short)0, cacc[tt], false, false);
        }
    }
    if (k0 < CCOL) {                       // K tail (peeled once, guarded loads)
        proj_stage<TM, true>(mat, row0, k0, tid, aTile[p]);
        __syncthreads();
        v16h b = *(const v16h*)(bcol + k0);
#pragma unroll
        for (int tt = 0; tt < TM; ++tt) {
            v16h a = *(const v16h*)(aTile[p] + tt * 512 + lane * 16);
            acc[tt]  = __builtin_amdgcn_wmma_f32_16x16x32_f16(false, a, false, b,
                                                              (short)0, acc[tt],  false, false);
            cacc[tt] = __builtin_amdgcn_wmma_f32_16x16x32_f16(false, a, false, onesb,
                                                              (short)0, cacc[tt], false, false);
        }
    }

#pragma unroll
    for (int tt = 0; tt < TM; ++tt) {
#pragma unroll
        for (int r = 0; r < 8; ++r) {
            int grow = row0 + tt * 16 + r + hi * 8;
            float c = cacc[tt][r];                     // exact nonzero count
            float val = (c > 0.5f) ? (acc[tt][r] / c) : 0.0f;
            emb[(size_t)grow * DD + col0 + m] = val;
        }
    }
}

// ---------------------------------------------------------------------------
// GCN pieces
// ---------------------------------------------------------------------------
__global__ __launch_bounds__(256)
void zero_kernel(float4* __restrict__ p, long long n4)
{
    long long i = (long long)blockIdx.x * 256 + threadIdx.x;
    if (i < n4) p[i] = make_float4(0.f, 0.f, 0.f, 0.f);
}

// One wave per edge: lane l handles dims [4l,4l+4) via one b128 load + 4 atomics.
__global__ __launch_bounds__(256)
void edge_scatter_kernel(const int* __restrict__ ei, const float* __restrict__ ea,
                         const float* __restrict__ h, float* __restrict__ agg, int E)
{
    int wave = threadIdx.x >> 5;
    int lane = threadIdx.x & 31;
    int e = blockIdx.x * 8 + wave;
    if (e >= E) return;
    int r = ei[e];
    int c = ei[E + e];
    float w = ea[e];
    float4 v = ((const float4*)(h + (size_t)r * DD))[lane];
    float* dst = agg + (size_t)c * DD + lane * 4;
    atomicAdd(dst + 0, v.x * w);
    atomicAdd(dst + 1, v.y * w);
    atomicAdd(dst + 2, v.z * w);
    atomicAdd(dst + 3, v.w * w);
}

// h_next = relu(agg[N,128] @ W[128,128]); TM row tiles share each B fragment.
template <int TM>
__global__ __launch_bounds__(256)
void gcn_gemm_kernel(const float* __restrict__ A, const _Float16* __restrict__ Wt,
                     float* __restrict__ dst)
{
    int wave = threadIdx.x >> 5;
    int lane = threadIdx.x & 31;
    int row0 = blockIdx.x * (16 * TM);
    int col0 = wave * 16;
    int m    = lane & 15;
    int hi   = lane >> 4;
    const _Float16* bcol = Wt + (size_t)(col0 + m) * DD + hi * 16;

    v8f acc[TM] = {};
#pragma unroll
    for (int k0 = 0; k0 < DD; k0 += 32) {
        v16h b = *(const v16h*)(bcol + k0);
        int kA = k0 + hi * 16;
#pragma unroll
        for (int tt = 0; tt < TM; ++tt) {
            const float* rowp = A + (size_t)(row0 + tt * 16 + m) * DD;
            float4 x0 = *(const float4*)(rowp + kA);
            float4 x1 = *(const float4*)(rowp + kA + 4);
            float4 y0 = *(const float4*)(rowp + kA + 8);
            float4 y1 = *(const float4*)(rowp + kA + 12);
            v16h a;
            a[0]  = (_Float16)x0.x; a[1]  = (_Float16)x0.y;
            a[2]  = (_Float16)x0.z; a[3]  = (_Float16)x0.w;
            a[4]  = (_Float16)x1.x; a[5]  = (_Float16)x1.y;
            a[6]  = (_Float16)x1.z; a[7]  = (_Float16)x1.w;
            a[8]  = (_Float16)y0.x; a[9]  = (_Float16)y0.y;
            a[10] = (_Float16)y0.z; a[11] = (_Float16)y0.w;
            a[12] = (_Float16)y1.x; a[13] = (_Float16)y1.y;
            a[14] = (_Float16)y1.z; a[15] = (_Float16)y1.w;
            acc[tt] = __builtin_amdgcn_wmma_f32_16x16x32_f16(false, a, false, b,
                                                             (short)0, acc[tt], false, false);
        }
    }
#pragma unroll
    for (int tt = 0; tt < TM; ++tt) {
#pragma unroll
        for (int r = 0; r < 8; ++r) {
            int grow = row0 + tt * 16 + r + hi * 8;
            float v = acc[tt][r];
            dst[(size_t)grow * DD + col0 + m] = v > 0.0f ? v : 0.0f;
        }
    }
}

// ---------------------------------------------------------------------------
// Metapath attention: one wave per node, P in {2,3}.
// ---------------------------------------------------------------------------
__global__ __launch_bounds__(256)
void att_kernel(const float* __restrict__ e0, const float* __restrict__ e1,
                const float* __restrict__ e2, int P,
                const float* __restrict__ wv, const float* __restrict__ bv,
                float* __restrict__ out, int N)
{
    int wave = threadIdx.x >> 5;
    int lane = threadIdx.x & 31;
    int n = blockIdx.x * 8 + wave;
    if (n >= N) return;
    const float* embs[3] = { e0, e1, e2 };
    float b = bv[0];
    float s[3];
    for (int p = 0; p < P; ++p) {
        const float* ep = embs[p] + (size_t)n * DD;
        float part = 0.0f;
#pragma unroll
        for (int q = 0; q < 4; ++q) part += ep[lane + 32 * q] * wv[lane + 32 * q];
#pragma unroll
        for (int off = 16; off > 0; off >>= 1) part += __shfl_xor(part, off, 32);
        s[p] = part + b;
    }
    float mx = s[0];
    for (int p = 1; p < P; ++p) mx = s[p] > mx ? s[p] : mx;
    float a[3]; float den = 0.0f;
    for (int p = 0; p < P; ++p) { a[p] = __expf(s[p] - mx); den += a[p]; }
    float inv = 1.0f / den;
#pragma unroll
    for (int q = 0; q < 4; ++q) {
        int d = lane + 32 * q;
        float v = 0.0f;
        for (int p = 0; p < P; ++p) v += a[p] * inv * embs[p][(size_t)n * DD + d];
        out[(size_t)n * DD + d] = v;
    }
}

// ---------------------------------------------------------------------------
// Host-side driver
// ---------------------------------------------------------------------------
template <int TM>
static void run_gcn(const float* emb, const int* ei, const float* ea, int E,
                    const _Float16* Wt, float* dst, int N,
                    float* agg, float* hA, float* hB, hipStream_t stream)
{
    long long n4 = (long long)N * DD / 4;
    int zgrid = (int)((n4 + 255) / 256);
    int egrid = (E + 7) / 8;
    int ggrid = N / (16 * TM);
    const float* src = emb;
    float* outs[3] = { hA, hB, dst };
    for (int l = 0; l < 3; ++l) {
        zero_kernel<<<zgrid, 256, 0, stream>>>((float4*)agg, n4);
        edge_scatter_kernel<<<egrid, 256, 0, stream>>>(ei, ea, src, agg, E);
        gcn_gemm_kernel<TM><<<ggrid, 256, 0, stream>>>(agg, Wt, outs[l]);
        src = outs[l];
    }
}

extern "C" void kernel_launch(void* const* d_in, const int* in_sizes, int n_in,
                              void* d_out, int out_size, void* d_ws, size_t ws_size,
                              hipStream_t stream)
{
    (void)in_sizes; (void)n_in; (void)out_size; (void)ws_size;

    const float* lrn_init = (const float*)d_in[0];
    const float* scn_init = (const float*)d_in[1];
    const float* cpt_init = (const float*)d_in[2];
    const int*   ei_lsl = (const int*)d_in[3];  const float* ea_lsl = (const float*)d_in[4];
    const int*   ei_scs = (const int*)d_in[5];  const float* ea_scs = (const float*)d_in[6];
    const int*   ei_sls = (const int*)d_in[7];  const float* ea_sls = (const float*)d_in[8];
    const int*   ei_cc  = (const int*)d_in[9];  const float* ea_cc  = (const float*)d_in[10];
    const int*   ei_cac = (const int*)d_in[11]; const float* ea_cac = (const float*)d_in[12];
    const int*   ei_csc = (const int*)d_in[13]; const float* ea_csc = (const float*)d_in[14];
    const float* pl_W1 = (const float*)d_in[15]; const float* pl_b1 = (const float*)d_in[16];
    const float* pl_W2 = (const float*)d_in[17]; const float* pl_b2 = (const float*)d_in[18];
    const float* ps_W1 = (const float*)d_in[19]; const float* ps_b1 = (const float*)d_in[20];
    const float* ps_W2 = (const float*)d_in[21]; const float* ps_b2 = (const float*)d_in[22];
    const float* pc_W1 = (const float*)d_in[23]; const float* pc_b1 = (const float*)d_in[24];
    const float* pc_W2 = (const float*)d_in[25]; const float* pc_b2 = (const float*)d_in[26];
    const float* W_lsl = (const float*)d_in[27];
    const float* W_cc  = (const float*)d_in[28];
    const float* W_cac = (const float*)d_in[29];
    const float* W_csc = (const float*)d_in[30];
    const float* W_scs = (const float*)d_in[31];
    const float* W_sls = (const float*)d_in[32];
    const float* att_cpt_w = (const float*)d_in[33];
    const float* att_cpt_b = (const float*)d_in[34];
    const float* att_scn_w = (const float*)d_in[35];
    const float* att_scn_b = (const float*)d_in[36];

    // ---- workspace carve (256B aligned) ----
    size_t off = 0;
    auto take = [&](size_t bytes) -> void* {
        void* p = (char*)d_ws + off;
        off += (bytes + 255) & ~(size_t)255;
        return p;
    };
    _Float16* tabLrn = (_Float16*)take((size_t)DD * CPAD * 2);
    _Float16* tabScn = (_Float16*)take((size_t)DD * CPAD * 2);
    _Float16* tabCpt = (_Float16*)take((size_t)DD * CPAD * 2);
    _Float16* Wt_lsl = (_Float16*)take((size_t)DD * DD * 2);
    _Float16* Wt_cc  = (_Float16*)take((size_t)DD * DD * 2);
    _Float16* Wt_cac = (_Float16*)take((size_t)DD * DD * 2);
    _Float16* Wt_csc = (_Float16*)take((size_t)DD * DD * 2);
    _Float16* Wt_scs = (_Float16*)take((size_t)DD * DD * 2);
    _Float16* Wt_sls = (_Float16*)take((size_t)DD * DD * 2);
    float* embLrn = (float*)take((size_t)NLRN * DD * 4);
    float* embScn = (float*)take((size_t)NSCN * DD * 4);
    float* embCpt = (float*)take((size_t)NCPT * DD * 4);
    float* hA     = (float*)take((size_t)NLRN * DD * 4);
    float* hB     = (float*)take((size_t)NLRN * DD * 4);
    float* agg    = (float*)take((size_t)NLRN * DD * 4);
    float* out_scs = (float*)take((size_t)NSCN * DD * 4);
    float* out_sls = (float*)take((size_t)NSCN * DD * 4);
    float* out_cc  = (float*)take((size_t)NCPT * DD * 4);
    float* out_cac = (float*)take((size_t)NCPT * DD * 4);
    float* out_csc = (float*)take((size_t)NCPT * DD * 4);

    float* out_lsl = (float*)d_out;
    float* fin_scn = (float*)d_out + (size_t)NLRN * DD;
    float* fin_cpt = (float*)d_out + (size_t)(NLRN + NSCN) * DD;

    // ---- 1) column tables + transposed f16 weights ----
    build_table_kernel<<<CPAD, 128, 0, stream>>>(pl_W1, pl_b1, pl_W2, pl_b2, tabLrn);
    build_table_kernel<<<CPAD, 128, 0, stream>>>(ps_W1, ps_b1, ps_W2, ps_b2, tabScn);
    build_table_kernel<<<CPAD, 128, 0, stream>>>(pc_W1, pc_b1, pc_W2, pc_b2, tabCpt);

    wt_kernel<<<DD, DD, 0, stream>>>(W_lsl, Wt_lsl);
    wt_kernel<<<DD, DD, 0, stream>>>(W_cc,  Wt_cc);
    wt_kernel<<<DD, DD, 0, stream>>>(W_cac, Wt_cac);
    wt_kernel<<<DD, DD, 0, stream>>>(W_csc, Wt_csc);
    wt_kernel<<<DD, DD, 0, stream>>>(W_scs, Wt_scs);
    wt_kernel<<<DD, DD, 0, stream>>>(W_sls, Wt_sls);

    // ---- 2) projection GEMMs (WMMA, double-buffered LDS A, fused masked-mean) ----
    proj_gemm_kernel<2><<<NLRN / 32, 256, 0, stream>>>(lrn_init, tabLrn, embLrn);
    proj_gemm_kernel<1><<<NSCN / 16, 256, 0, stream>>>(scn_init, tabScn, embScn);
    proj_gemm_kernel<2><<<NCPT / 32, 256, 0, stream>>>(cpt_init, tabCpt, embCpt);

    // ---- 3) six 3-layer GCNs ----
    run_gcn<2>(embLrn, ei_lsl, ea_lsl, 1000000, Wt_lsl, out_lsl, NLRN, agg, hA, hB, stream);
    run_gcn<1>(embScn, ei_scs, ea_scs, 200000,  Wt_scs, out_scs, NSCN, agg, hA, hB, stream);
    run_gcn<1>(embScn, ei_sls, ea_sls, 200000,  Wt_sls, out_sls, NSCN, agg, hA, hB, stream);
    run_gcn<2>(embCpt, ei_cc,  ea_cc,  500000,  Wt_cc,  out_cc,  NCPT, agg, hA, hB, stream);
    run_gcn<2>(embCpt, ei_cac, ea_cac, 500000,  Wt_cac, out_cac, NCPT, agg, hA, hB, stream);
    run_gcn<2>(embCpt, ei_csc, ea_csc, 500000,  Wt_csc, out_csc, NCPT, agg, hA, hB, stream);

    // ---- 4) metapath attention ----
    att_kernel<<<(NSCN + 7) / 8, 256, 0, stream>>>(out_scs, out_sls, out_sls, 2,
                                                   att_scn_w, att_scn_b, fin_scn, NSCN);
    att_kernel<<<(NCPT + 7) / 8, 256, 0, stream>>>(out_cc, out_cac, out_csc, 3,
                                                   att_cpt_w, att_cpt_b, fin_cpt, NCPT);
}
